// ARNN_5652176961890
// MI455X (gfx1250) — compile-verified, matrix operand order checked by reading, not run on hardware
//
#include <hip/hip_runtime.h>
#include <stdint.h>

#define Bn 4096
#define Sn 512
#define In 12
#define En 8
#define Hn 32

typedef _Float16 half_t;
typedef _Float16 v16h __attribute__((ext_vector_type(16)));
typedef _Float16 h8   __attribute__((ext_vector_type(8)));
typedef float    v8f  __attribute__((ext_vector_type(8)));
typedef unsigned int u32x4 __attribute__((ext_vector_type(4)));
typedef int          i32x8 __attribute__((ext_vector_type(8)));
typedef int          i32x4 __attribute__((ext_vector_type(4)));

#if __has_builtin(__builtin_amdgcn_tensor_load_to_lds)
#define HAVE_TDM 1
#else
#define HAVE_TDM 0
#endif

__device__ __forceinline__ v8f splat8(float x){
  v8f v;
#pragma unroll
  for(int i=0;i<8;i++) v[i]=x;
  return v;
}
__device__ __forceinline__ h8 zh8(){
  h8 v;
#pragma unroll
  for(int i=0;i<8;i++) v[i]=(half_t)0.f;
  return v;
}
__device__ __forceinline__ v16h frag16(h8 lo, h8 hi){
  return __builtin_shufflevector(lo,hi,0,1,2,3,4,5,6,7,8,9,10,11,12,13,14,15);
}
// Build a 16-bit WMMA operand fragment from a row-major [16+][>=32] LDS tile.
// A-matrix: row index = lane%16 = M.  B-matrix: stage W^T as [N][K]; row = lane%16 = N.
// Per ISA 16-bit 16x32 layout: lanes 0-15 hold K 0..7 (v0-3) and 16..23 (v4-7);
// lanes 16-31 hold K 8..15 and 24..31.
__device__ __forceinline__ v16h ldsfrag(const half_t* base,int stride,int lane){
  int r=lane&15, hb=(lane>>4)&1;
  const h8* p0=(const h8*)(base + r*stride + hb*8);
  const h8* p1=(const h8*)(base + r*stride + 16 + hb*8);
  return frag16(*p0,*p1);
}
__device__ __forceinline__ v8f wmma16(v16h a, v16h b, v8f c){
  return __builtin_amdgcn_wmma_f32_16x16x32_f16(false,a,false,b,(short)0,c,false,false);
}
// fast math: v_rcp_f32 instead of IEEE division sequences
__device__ __forceinline__ float rcp_(float x){ return __builtin_amdgcn_rcpf(x); }
__device__ __forceinline__ float sigm(float x){ return rcp_(1.0f+__expf(-x)); }
__device__ __forceinline__ float tanh_(float x){ float e=__expf(-2.0f*x); return (1.0f-e)*rcp_(1.0f+e); }
__device__ __forceinline__ float lrelu(float x){ return x>0.f?x:0.01f*x; }

#if HAVE_TDM
// 1-D TDM copy: 1KB contiguous tile (512 x 2-byte elements) global -> LDS.
// D# per CDNA5 ISA 8.3-8.6: group0 = {count=1 | lds_addr | global_addr | type=2},
// group1 = {data_size=1(2B), tensor_dim0=512, tensor_dim1=1, tile_dim0=512,
//           tile_dim1/2=0(unused), tensor_dim0_stride=512}.
// This toolchain exposes the 6-arg builtin: (u32x4, i32x8, i32x4, i32x4, i32x8, cpol).
__device__ __forceinline__ void tdm_load_tile(const half_t* gsrc, half_t* ldst){
  uint64_t ga=(uint64_t)(uintptr_t)gsrc;
  uint32_t la=(uint32_t)(uintptr_t)ldst;     // low 32 bits of shared aperture = LDS offset
  u32x4 g0 = { 1u, la, (unsigned)ga, (unsigned)(ga>>32) | (2u<<30) };
  i32x8 g1 = { 0x10000, (int)(512u<<16), (int)(1u<<16), (int)(512u<<16), 0, 512, 0, 0 };
  i32x4 gz = { 0,0,0,0 };
  i32x8 gz8 = { 0,0,0,0,0,0,0,0 };
  __builtin_amdgcn_tensor_load_to_lds(g0,g1,gz,gz,gz8,0);
}
#endif

// -------------------------------------------------------------------------
// Kernel 1: v[s][b][e] = lrelu(x[b][s][:] @ We + be), stored fp16 (S,B,E)
// -------------------------------------------------------------------------
__global__ __launch_bounds__(256) void embed_kernel(
    const float* __restrict__ x, const float* __restrict__ We,
    const float* __restrict__ be, half_t* __restrict__ v)
{
  __shared__ float sWe[In*En];
  __shared__ float sbe[En];
  __shared__ alignas(16) half_t tile[16][16][En]; // [si][bi][e]
  int tid=threadIdx.x;
  if(tid<In*En) sWe[tid]=We[tid];
  if(tid<En)    sbe[tid]=be[tid];
  __syncthreads();
  int bt=blockIdx.x%(Bn/16);
  int st=blockIdx.x/(Bn/16);
  int bi=tid>>4, si=tid&15;
  const float* xp=x + ((size_t)(bt*16+bi)*Sn + st*16 + si)*In;
  float xv[In];
#pragma unroll
  for(int i=0;i<In;i++) xv[i]=xp[i];
#pragma unroll
  for(int e=0;e<En;e++){
    float acc=sbe[e];
#pragma unroll
    for(int i=0;i<In;i++) acc+=xv[i]*sWe[i*En+e];
    tile[si][bi][e]=(half_t)lrelu(acc);
  }
  __syncthreads();
  int si2=tid>>4, bi2=tid&15;
  h8* dst=(h8*)(v + ((size_t)(st*16+si2)*Bn + bt*16 + bi2)*En);
  *dst=*(const h8*)&tile[si2][bi2][0];
}

// -------------------------------------------------------------------------
// Kernel 2: LSTM. One wave per 16-batch tile. feat K-order: [v(8)|h(32)|0(24)].
// 16 v_wmma_f32_16x16x32_f16 per timestep; hs writeback via
// global_store_async_from_lds_b128 (async engine, ASYNCcnt-fenced).
// -------------------------------------------------------------------------
__global__ __launch_bounds__(128,1) void lstm_kernel(
    const half_t* __restrict__ v, const float* __restrict__ W_ih,
    const float* __restrict__ W_hh, const float* __restrict__ b_ih,
    const float* __restrict__ b_hh, half_t* __restrict__ hs)
{
  __shared__ alignas(16) half_t Wcat[128][64];
  __shared__ alignas(16) half_t hstage[4][16][32];
  __shared__ alignas(16) half_t zpad[16];
  int tid=threadIdx.x;
  {
    int n=tid;
#pragma unroll
    for(int k=0;k<8;k++)   Wcat[n][k]   =(half_t)W_ih[n*En+k];
#pragma unroll
    for(int k=0;k<32;k++)  Wcat[n][8+k] =(half_t)W_hh[n*Hn+k];
#pragma unroll
    for(int k=40;k<64;k++) Wcat[n][k]   =(half_t)0.f;
  }
  for(int i=tid;i<4*16*32;i+=128) ((half_t*)hstage)[i]=(half_t)0.f;
  if(tid<16) zpad[tid]=(half_t)0.f;
  __syncthreads();

  int lane=tid&31, w=tid>>5;
  int r=lane&15, hb=lane>>4;
  int b0=(blockIdx.x*4+w)*16;

  v16h Bf[8][2];
#pragma unroll
  for(int t=0;t<8;t++)
#pragma unroll
    for(int kf=0;kf<2;kf++)
      Bf[t][kf]=ldsfrag(&Wcat[t*16][0]+kf*32,64,lane);

  float bias[8];
#pragma unroll
  for(int t=0;t<8;t++){ int n=t*16+r; bias[t]=b_ih[n]+b_hh[n]; }

  v8f cst0=splat8(0.f), cst1=splat8(0.f);
  half_t* hst=&hstage[w][0][0];
  const h8 z8=zh8();

  for(int s=0;s<Sn;s++){
    const h8* vg=(const h8*)(v + ((size_t)s*Bn + b0 + r)*En);
    h8 vrow=*vg;                                   // lanes 0-15: A frag0 K0-7 directly
    if(s+1<Sn) __builtin_prefetch(v + ((size_t)(s+1)*Bn + b0 + r)*En, 0, 0);
    h8 dlo=*(const h8*)(hst + r*32);               // h[r][0:8]
    h8 dhi=*(const h8*)(hst + r*32 + (hb?16:8));   // h[r][16:24] / h[r][8:16]
    const half_t* p1 = hb ? &zpad[0] : (hst + r*32 + 24);
    h8 d1=*(const h8*)p1;                          // h[r][24:32] or zeros
    v16h a0=frag16(hb?dlo:vrow, dhi);
    v16h a1=frag16(d1, z8);

    v8f g[8];
#pragma unroll
    for(int t=0;t<8;t++){
      v8f acc=splat8(bias[t]);
      acc=wmma16(a0,Bf[t][0],acc);
      g[t]=wmma16(a1,Bf[t][1],acc);
    }
    // tiles 0,1=i  2,3=f  4,5=g  6,7=o ; lane holds m=e+8*hb, j=u*16+r
    float hnew[2][8];
#pragma unroll
    for(int e=0;e<8;e++){
      float i0=sigm(g[0][e]), f0=sigm(g[2][e]), g0=tanh_(g[4][e]), o0=sigm(g[6][e]);
      float c0=f0*cst0[e]+i0*g0; cst0[e]=c0; hnew[0][e]=o0*tanh_(c0);
      float i1=sigm(g[1][e]), f1=sigm(g[3][e]), g1=tanh_(g[5][e]), o1=sigm(g[7][e]);
      float c1=f1*cst1[e]+i1*g1; cst1[e]=c1; hnew[1][e]=o1*tanh_(c1);
    }
    // previous iteration's async stores must have drained before we overwrite hstage
    asm volatile("s_wait_asynccnt 0x0" ::: "memory");
#pragma unroll
    for(int e=0;e<8;e++){
      hst[(e+8*hb)*32 + r]      =(half_t)hnew[0][e];
      hst[(e+8*hb)*32 + 16 + r] =(half_t)hnew[1][e];
    }
    // async LDS->global writeback of the contiguous 1KB tile (hs[s][b0:b0+16][:])
    {
      uint64_t ga=(uint64_t)(uintptr_t)(hs + ((size_t)s*Bn + b0)*Hn) + (uint32_t)lane*16u;
      uint32_t la=(uint32_t)(uintptr_t)hst + (uint32_t)lane*16u;
      asm volatile("s_wait_dscnt 0x0\n\t"
                   "global_store_async_from_lds_b128 %0, %1, off\n\t"
                   "global_store_async_from_lds_b128 %0, %1, off offset:512"
                   :: "v"(ga), "v"(la) : "memory");
    }
  }
}

// -------------------------------------------------------------------------
// Kernel 3: attention MLP (40->16->32->1, tanh) with ONLINE softmax
// (logits in [-1,1] => no max tracking), h accumulated in registers, then
// regressor MLP (40->32->64->1, sigmoid). hs tiles arrive via TDM
// (tensor_load_to_lds, double-buffered) when available.
// -------------------------------------------------------------------------
__global__ __launch_bounds__(128,1) void attn_kernel(
  const half_t* __restrict__ v, const half_t* __restrict__ hs,
  const float* __restrict__ Ew1,const float* __restrict__ Eb1,
  const float* __restrict__ Ew2,const float* __restrict__ Eb2,
  const float* __restrict__ Ew3,const float* __restrict__ Eb3,
  const float* __restrict__ Rw1,const float* __restrict__ Rb1,
  const float* __restrict__ Rw2,const float* __restrict__ Rb2,
  const float* __restrict__ Rw3,const float* __restrict__ Rb3,
  float* __restrict__ out)
{
  __shared__ alignas(16) half_t W1T[16][64];
  __shared__ alignas(16) half_t W2T[32][32];
  __shared__ alignas(16) half_t W3T[16][32];
  __shared__ alignas(16) half_t R1T[32][64];
  __shared__ alignas(16) half_t R2T[64][32];
  __shared__ alignas(16) half_t R3T[16][64];
  __shared__ alignas(16) half_t st1[4][16][32];
  __shared__ alignas(16) half_t st2[4][16][64];
#if HAVE_TDM
  __shared__ alignas(16) half_t tbuf[4][2][16*32];
#endif
  __shared__ float est[4][16];
  int tid=threadIdx.x;
  for(int i=tid;i<16*64;i+=128){int n=i>>6,k=i&63; W1T[n][k]=(half_t)(k<40?Ew1[k*16+n]:0.f);}
  for(int i=tid;i<32*32;i+=128){int n=i>>5,k=i&31; W2T[n][k]=(half_t)(k<16?Ew2[k*32+n]:0.f);}
  for(int i=tid;i<16*32;i+=128){int n=i>>5,k=i&31; W3T[n][k]=(half_t)(n==0?Ew3[k]:0.f);}
  for(int i=tid;i<32*64;i+=128){int n=i>>6,k=i&63; R1T[n][k]=(half_t)(k<40?Rw1[k*32+n]:0.f);}
  for(int i=tid;i<64*32;i+=128){int n=i>>5,k=i&31; R2T[n][k]=(half_t)Rw2[k*64+n];}
  for(int i=tid;i<16*64;i+=128){int n=i>>6,k=i&63; R3T[n][k]=(half_t)(n==0?Rw3[k]:0.f);}
  for(int i=tid;i<4*16*32;i+=128) ((half_t*)st1)[i]=(half_t)0.f;
  for(int i=tid;i<4*16*64;i+=128) ((half_t*)st2)[i]=(half_t)0.f;
  __syncthreads();

  int lane=tid&31, w=tid>>5;
  int r=lane&15, hb=lane>>4;
  int b0=(blockIdx.x*4+w)*16;

  v16h W1f0=ldsfrag(&W1T[0][0],64,lane);
  v16h W1f1=ldsfrag(&W1T[0][0]+32,64,lane);
  v16h W2f0=ldsfrag(&W2T[0][0],32,lane);
  v16h W2f1=ldsfrag(&W2T[16][0],32,lane);
  v16h W3f =ldsfrag(&W3T[0][0],32,lane);
  float eb1=Eb1[r], eb2a=Eb2[r], eb2b=Eb2[16+r], eb3=Eb3[0];

  const h8 z8=zh8();
  const h8* vl=(const h8*)(v + ((size_t)(Sn-1)*Bn + b0 + r)*En);
  h8 vrow=*vl;
  v16h fragV=frag16(hb? z8 : vrow, z8);   // v_last occupies feat K32..39

  half_t* s1=&st1[w][0][0];
  half_t* s2=&st2[w][0][0];
  float*  ew=&est[w][0];

  v8f haccL=splat8(0.f), haccH=splat8(0.f);
  float sume=0.f;

#if HAVE_TDM
  tdm_load_tile(hs + (size_t)b0*Hn, &tbuf[w][0][0]);   // s = 0
#endif
  for(int s=0;s<Sn;s++){
#if HAVE_TDM
    __builtin_amdgcn_s_wait_tensorcnt(0);              // tile s resident
    if(s+1<Sn) tdm_load_tile(hs + ((size_t)(s+1)*Bn + b0)*Hn, &tbuf[w][(s+1)&1][0]);
    const half_t* hp=&tbuf[w][s&1][0] + r*Hn;
#else
    const half_t* hp=hs + ((size_t)s*Bn + b0 + r)*Hn;
    if(s+1<Sn) __builtin_prefetch(hs + ((size_t)(s+1)*Bn + b0 + r)*Hn, 0, 0);
#endif
    h8 hlo=*(const h8*)(hp + hb*8);        // lane's h[r][0:8] / [8:16]
    h8 hhi=*(const h8*)(hp + 16 + hb*8);   // h[r][16:24] / [24:32]
    v16h fragH=frag16(hlo,hhi);
    v8f l1=wmma16(fragH,W1f0,splat8(eb1));
    l1=wmma16(fragV,W1f1,l1);
#pragma unroll
    for(int e=0;e<8;e++) s1[(e+8*hb)*32 + r]=(half_t)lrelu(l1[e]);
    v16h a2=ldsfrag(s1,32,lane);
    v8f l2a=wmma16(a2,W2f0,splat8(eb2a));
    v8f l2b=wmma16(a2,W2f1,splat8(eb2b));
#pragma unroll
    for(int e=0;e<8;e++){
      s2[(e+8*hb)*64 + r]    =(half_t)lrelu(l2a[e]);
      s2[(e+8*hb)*64 + 16+r] =(half_t)lrelu(l2b[e]);
    }
    v16h a3=ldsfrag(s2,64,lane);
    v8f l3=wmma16(a3,W3f,splat8(eb3));
    if(r==0){                              // lanes 0,16 hold the n==0 column
#pragma unroll
      for(int e=0;e<8;e++) ew[e+8*hb]=__expf(tanh_(l3[e]));
    }
    float ee=ew[r];                        // broadcast exp(logit) for this lane's m
    sume+=ee;
#pragma unroll
    for(int e=0;e<8;e++){ haccL[e]+=ee*(float)hlo[e]; haccH[e]+=ee*(float)hhi[e]; }
  }

  float rs=rcp_(sume);
  h8 hfl, hfh;
#pragma unroll
  for(int e=0;e<8;e++){ hfl[e]=(half_t)(haccL[e]*rs); hfh[e]=(half_t)(haccH[e]*rs); }
  v16h fragF=frag16(hfl,hfh);              // h_final already lane-placed as A frag0

  v16h R1f00=ldsfrag(&R1T[0][0],64,lane);
  v16h R1f01=ldsfrag(&R1T[0][0]+32,64,lane);
  v16h R1f10=ldsfrag(&R1T[16][0],64,lane);
  v16h R1f11=ldsfrag(&R1T[16][0]+32,64,lane);
  v8f m1a=wmma16(fragV,R1f01, wmma16(fragF,R1f00,splat8(Rb1[r])));
  v8f m1b=wmma16(fragV,R1f11, wmma16(fragF,R1f10,splat8(Rb1[16+r])));
#pragma unroll
  for(int e=0;e<8;e++){
    s2[(e+8*hb)*64 + r]   =(half_t)lrelu(m1a[e]);
    s2[(e+8*hb)*64 +16+r] =(half_t)lrelu(m1b[e]);
  }
  v16h aR2=ldsfrag(s2,64,lane);
  v8f m2[4];
#pragma unroll
  for(int t=0;t<4;t++){
    v16h R2f=ldsfrag(&R2T[t*16][0],32,lane);
    m2[t]=wmma16(aR2,R2f,splat8(Rb2[t*16+r]));
  }
#pragma unroll
  for(int t=0;t<4;t++)
#pragma unroll
    for(int e=0;e<8;e++) s2[(e+8*hb)*64 + t*16 + r]=(half_t)lrelu(m2[t][e]);
  v16h aR3a=ldsfrag(s2,64,lane);
  v16h aR3b=ldsfrag(s2+32,64,lane);
  v16h R3f0=ldsfrag(&R3T[0][0],64,lane);
  v16h R3f1=ldsfrag(&R3T[0][0]+32,64,lane);
  v8f m3=wmma16(aR3b,R3f1, wmma16(aR3a,R3f0,splat8(Rb3[0])));
  if(r==0){
#pragma unroll
    for(int e=0;e<8;e++) out[b0 + e + 8*hb]=rcp_(1.0f+__expf(-m3[e]));
  }
}

extern "C" void kernel_launch(void* const* d_in, const int* in_sizes, int n_in,
                              void* d_out, int out_size, void* d_ws, size_t ws_size,
                              hipStream_t stream){
  const float* x   =(const float*)d_in[0];
  const float* We  =(const float*)d_in[1];
  const float* be  =(const float*)d_in[2];
  const float* W_ih=(const float*)d_in[3];
  const float* W_hh=(const float*)d_in[4];
  const float* b_ih=(const float*)d_in[5];
  const float* b_hh=(const float*)d_in[6];
  const float* Ew1 =(const float*)d_in[7];
  const float* Eb1 =(const float*)d_in[8];
  const float* Ew2 =(const float*)d_in[9];
  const float* Eb2 =(const float*)d_in[10];
  const float* Ew3 =(const float*)d_in[11];
  const float* Eb3 =(const float*)d_in[12];
  const float* Rw1 =(const float*)d_in[13];
  const float* Rb1 =(const float*)d_in[14];
  const float* Rw2 =(const float*)d_in[15];
  const float* Rb2 =(const float*)d_in[16];
  const float* Rw3 =(const float*)d_in[17];
  const float* Rb3 =(const float*)d_in[18];
  half_t* vbuf =(half_t*)d_ws;                         // (S,B,E) fp16 : 32 MB
  half_t* hsbuf=vbuf + (size_t)Sn*Bn*En;               // (S,B,H) fp16 : 128 MB
  embed_kernel<<<dim3((Bn/16)*(Sn/16)),dim3(256),0,stream>>>(x,We,be,vbuf);
  lstm_kernel <<<dim3(Bn/64),dim3(128),0,stream>>>(vbuf,W_ih,W_hh,b_ih,b_hh,hsbuf);
  attn_kernel <<<dim3(Bn/64),dim3(128),0,stream>>>(vbuf,hsbuf,Ew1,Eb1,Ew2,Eb2,Ew3,Eb3,
                                                   Rw1,Rb1,Rw2,Rb2,Rw3,Rb3,(float*)d_out);
}
